// Generator_AE_82085414961894
// MI455X (gfx1250) — matrix operand order repaired and checked by reading, not compiled, not run on hardware
//
#include <hip/hip_runtime.h>
#include <hip/hip_bf16.h>
#include <cstdint>
#include <cstddef>

typedef __attribute__((ext_vector_type(16))) _Float16 v16h;
typedef __attribute__((ext_vector_type(8)))  _Float16 v8h;
typedef __attribute__((ext_vector_type(8)))  float    v8f;
typedef __attribute__((ext_vector_type(4)))  float    v4f;

#define BATCH 50

// ---------------- encoder conv: 3x3, stride 2, pad 1, bias + leaky ----------------
__global__ void conv3x3_s2p1_leaky(const float* __restrict__ in,
                                   const float* __restrict__ w,
                                   const float* __restrict__ bias,
                                   float* __restrict__ out,
                                   int IC, int IH, int IW,
                                   int OC, int OH, int OW)
{
    int idx = blockIdx.x * blockDim.x + threadIdx.x;
    int total = BATCH * OC * OH * OW;
    if (idx >= total) return;
    int ox = idx % OW;
    int oy = (idx / OW) % OH;
    int oc = (idx / (OW * OH)) % OC;
    int b  = idx / (OW * OH * OC);
    float s = bias[oc];
    int iy0 = oy * 2 - 1, ix0 = ox * 2 - 1;
    for (int ic = 0; ic < IC; ++ic) {
        const float* ip = in + ((size_t)(b * IC + ic)) * IH * IW;
        const float* wp = w + ((size_t)(oc * IC + ic)) * 9;
#pragma unroll
        for (int ky = 0; ky < 3; ++ky) {
            int iy = iy0 + ky;
            if ((unsigned)iy >= (unsigned)IH) continue;
#pragma unroll
            for (int kx = 0; kx < 3; ++kx) {
                int ix = ix0 + kx;
                if ((unsigned)ix >= (unsigned)IW) continue;
                s = fmaf(ip[iy * IW + ix], wp[ky * 3 + kx], s);
            }
        }
    }
    out[idx] = (s >= 0.f) ? s : 0.01f * s;
}

// ---------------- batchnorm stats (biased var, eps 1e-5) ----------------
__global__ void bn_stats(const float* __restrict__ x, float* __restrict__ mu,
                         float* __restrict__ rsig, int C, int HW)
{
    int c = blockIdx.x;
    __shared__ double ss[256], sq[256];
    double s = 0.0, q = 0.0;
    int n = BATCH * HW;
    for (int t = threadIdx.x; t < n; t += blockDim.x) {
        int b = t / HW, i = t - b * HW;
        float v = x[((size_t)b * C + c) * HW + i];
        s += (double)v;
        q += (double)v * (double)v;
    }
    ss[threadIdx.x] = s; sq[threadIdx.x] = q;
    __syncthreads();
    for (int st = blockDim.x / 2; st > 0; st >>= 1) {
        if ((int)threadIdx.x < st) {
            ss[threadIdx.x] += ss[threadIdx.x + st];
            sq[threadIdx.x] += sq[threadIdx.x + st];
        }
        __syncthreads();
    }
    if (threadIdx.x == 0) {
        double m   = ss[0] / n;
        double var = sq[0] / n - m * m;
        mu[c]   = (float)m;
        rsig[c] = rsqrtf((float)var + 1e-5f);
    }
}

__global__ void bn_apply(float* __restrict__ x, const float* __restrict__ mu,
                         const float* __restrict__ rsig, int C, int HW, int total)
{
    int idx = blockIdx.x * blockDim.x + threadIdx.x;
    if (idx >= total) return;
    int c = (idx / HW) % C;
    x[idx] = (x[idx] - mu[c]) * rsig[c];
}

// ---------------- pack f32 rows -> zero-padded 64-row f16 A matrix ----------------
__global__ void pack_rows_f16(const float* __restrict__ src, _Float16* __restrict__ dst,
                              int rows, int K)
{
    int idx = blockIdx.x * blockDim.x + threadIdx.x;
    int total = 64 * K;
    if (idx >= total) return;
    int r = idx / K;
    int k = idx - r * K;
    dst[idx] = (r < rows) ? (_Float16)src[(size_t)r * K + k] : (_Float16)0.f;
}

// ---------------- pack f32 rows -> zero-padded f32 B matrix (encoder we) ----------------
__global__ void pad_rows_f32(const float* __restrict__ src, float* __restrict__ dst,
                             int rows, int rows_pad, int K)
{
    int idx = blockIdx.x * blockDim.x + threadIdx.x;
    int total = rows_pad * K;
    if (idx >= total) return;
    int r = idx / K;
    int k = idx - r * K;
    dst[idx] = (r < rows) ? src[(size_t)r * K + k] : 0.f;
}

// ---------------- WMMA GEMM: D[m,n] = sum_k A[m,k]*B[n,k] ----------------
// A: [64][K] f16 row-major (zero-padded rows).  B: [>=gridDim.x*64][K] f32 row-major,
// ALL rows addressable (caller zero-pads).  Block = 256 threads = 8 waves =
// 4 m-tiles x 2 n-groups; each wave computes 16M x 32N (2 WMMA tiles, A reused).
// grid.x tiles N by 64, grid.y splits K into chunks of iters*32 -> partial sums.
__global__ __launch_bounds__(256)
void gemm_nt_wmma(const _Float16* __restrict__ A,
                  const float* __restrict__ Bm,
                  float* __restrict__ part,      // [gridDim.y][64][ldp]
                  int K, int ldp, int iters)
{
    int lane   = threadIdx.x & 31;
    int wave   = threadIdx.x >> 5;
    int m_tile = wave & 3;
    int n_half = wave >> 2;                 // 0 or 1
    int l15    = lane & 15;
    int hi     = lane >> 4;                 // 0 or 1
    int n_base = blockIdx.x * 64 + n_half * 32;
    int k0     = blockIdx.y * iters * 32;

    // A fragment: lane holds row m; K offset (hi?16:0), 16 consecutive halves
    const _Float16* Arow = A + (size_t)(m_tile * 16 + l15) * K + k0 + hi * 16;
    // B fragments: lane holds rows o and o+16; K offset (hi?16:0), 16 consecutive f32
    const float* B0 = Bm + (size_t)(n_base + l15) * K + k0 + hi * 16;
    const float* B1 = B0 + (size_t)16 * K;

    union AU { v16h v; v8h h[2]; };
    union BU { v16h v; _Float16 e[16]; };
    v8f acc0 = {}, acc1 = {};

    for (int it = 0; it < iters; ++it) {
        AU a;
        a.h[0] = *(const v8h*)(Arow);
        a.h[1] = *(const v8h*)(Arow + 8);
        Arow += 32;

        __builtin_prefetch(B0 + 256, 0, 1);   // stream-ahead on B (wd)
        __builtin_prefetch(B1 + 256, 0, 1);
        const v4f* p0 = (const v4f*)B0;
        const v4f* p1 = (const v4f*)B1;
        BU b0, b1;
#pragma unroll
        for (int j = 0; j < 4; ++j) {
            v4f t0 = p0[j];
            v4f t1 = p1[j];
            b0.e[4 * j + 0] = (_Float16)t0[0];
            b0.e[4 * j + 1] = (_Float16)t0[1];
            b0.e[4 * j + 2] = (_Float16)t0[2];
            b0.e[4 * j + 3] = (_Float16)t0[3];
            b1.e[4 * j + 0] = (_Float16)t1[0];
            b1.e[4 * j + 1] = (_Float16)t1[1];
            b1.e[4 * j + 2] = (_Float16)t1[2];
            b1.e[4 * j + 3] = (_Float16)t1[3];
        }
        B0 += 32;
        B1 += 32;

        acc0 = __builtin_amdgcn_wmma_f32_16x16x32_f16(
            false, a.v, false, b0.v, (short)0, acc0, false, false);
        acc1 = __builtin_amdgcn_wmma_f32_16x16x32_f16(
            false, a.v, false, b1.v, (short)0, acc1, false, false);
    }

    // D layout: lane l, vgpr r -> (m = r + 8*hi, n = l&15)
    float* pp = part + ((size_t)blockIdx.y * 64 + m_tile * 16 + hi * 8) * ldp + n_base + l15;
#pragma unroll
    for (int r = 0; r < 8; ++r) {
        pp[(size_t)r * ldp]      = acc0[r];
        pp[(size_t)r * ldp + 16] = acc1[r];
    }
}

// ---------------- reduce K-chunk partials + bias (+optional relu) ----------------
__global__ void fc_finalize(const float* __restrict__ part, const float* __restrict__ bias,
                            float* __restrict__ out, int N, int ldp, int KS, int do_relu)
{
    int idx = blockIdx.x * blockDim.x + threadIdx.x;
    if (idx >= BATCH * N) return;
    int b = idx / N, n = idx - b * N;
    float s = bias[n];
    for (int c = 0; c < KS; ++c) s += part[((size_t)c * 64 + b) * ldp + n];
    if (do_relu) s = fmaxf(s, 0.f);
    out[idx] = s;
}

// ---------------- v = sqrt(N) * real(ifft2(e.reshape(B,25,8))) ----------------
__global__ void ifft2_real(const float* __restrict__ e, float* __restrict__ v)
{
    int idx = blockIdx.x * blockDim.x + threadIdx.x;
    if (idx >= BATCH * 200) return;
    int b = idx / 200, i = idx - b * 200;
    int m = i >> 3, l = i & 7;
    const float TWO_PI = 6.28318530717958647692f;
    const float* eb = e + b * 200;
    float s = 0.f;
    for (int p = 0; p < 25; ++p) {
        int mp = (m * p) % 25;                       // periodicity reduction
        float am = TWO_PI * ((float)mp * (1.f / 25.f));
#pragma unroll
        for (int q = 0; q < 8; ++q) {
            int lq = (l * q) & 7;
            float ang = am + TWO_PI * ((float)lq * 0.125f);
            s = fmaf(eb[p * 8 + q], __cosf(ang), s);
        }
    }
    v[idx] = s * 0.07071067811865475f;               // 1/sqrt(200)
}

// ---------------- C[b,i,j] = v[b,(i-j)%200] -> d_out and f16 A for dec GEMM ----------------
__global__ void build_C_Adec(const float* __restrict__ v, _Float16* __restrict__ Adec,
                             float* __restrict__ Cout)
{
    int idx = blockIdx.x * blockDim.x + threadIdx.x;
    const int total = 64 * 40000;
    if (idx >= total) return;
    int b = idx / 40000;
    int k = idx - b * 40000;
    if (b < BATCH) {
        int i = k / 200, j = k - i * 200;
        int d = i - j; if (d < 0) d += 200;
        float val = v[b * 200 + d];
        Adec[idx] = (_Float16)val;
        Cout[(size_t)b * 40000 + k] = val;
    } else {
        Adec[idx] = (_Float16)0.f;
    }
}

// ---------------- transposed conv 3x3 (gather form), weight layout (IC,OC,3,3) ----------------
__global__ void convt3x3(const float* __restrict__ in, const float* __restrict__ w,
                         const float* __restrict__ bias, float* __restrict__ out,
                         int IC, int IH, int IW, int OC, int OH, int OW, int S, int P)
{
    int idx = blockIdx.x * blockDim.x + threadIdx.x;
    int total = BATCH * OC * OH * OW;
    if (idx >= total) return;
    int ox = idx % OW;
    int oy = (idx / OW) % OH;
    int oc = (idx / (OW * OH)) % OC;
    int b  = idx / (OW * OH * OC);
    float s = bias[oc];
    int ihw = IH * IW;
#pragma unroll
    for (int ky = 0; ky < 3; ++ky) {
        int t = oy + P - ky;
        if (t < 0 || (t % S) != 0) continue;
        int iy = t / S; if (iy >= IH) continue;
#pragma unroll
        for (int kx = 0; kx < 3; ++kx) {
            int u = ox + P - kx;
            if (u < 0 || (u % S) != 0) continue;
            int ix = u / S; if (ix >= IW) continue;
            const float* ip = in + (size_t)b * IC * ihw + iy * IW + ix;
            const float* wp = w + (size_t)oc * 9 + ky * 3 + kx;
            float acc = 0.f;
            for (int ic = 0; ic < IC; ++ic)
                acc = fmaf(ip[(size_t)ic * ihw], wp[(size_t)ic * OC * 9], acc);
            s += acc;
        }
    }
    out[idx] = s;
}

static inline int cdiv(int a, int b) { return (a + b - 1) / b; }

extern "C" void kernel_launch(void* const* d_in, const int* in_sizes, int n_in,
                              void* d_out, int out_size, void* d_ws, size_t ws_size,
                              hipStream_t stream) {
    const float* x   = (const float*)d_in[0];
    const float* cw1 = (const float*)d_in[1];  const float* cb1 = (const float*)d_in[2];
    const float* cw2 = (const float*)d_in[3];  const float* cb2 = (const float*)d_in[4];
    const float* cw3 = (const float*)d_in[5];  const float* cb3 = (const float*)d_in[6];
    const float* cw4 = (const float*)d_in[7];  const float* cb4 = (const float*)d_in[8];
    const float* we  = (const float*)d_in[9];  const float* be  = (const float*)d_in[10];
    const float* wd  = (const float*)d_in[11]; const float* bd  = (const float*)d_in[12];
    const float* tw1 = (const float*)d_in[13]; const float* tb1 = (const float*)d_in[14];
    const float* tw2 = (const float*)d_in[15]; const float* tb2 = (const float*)d_in[16];
    const float* tw3 = (const float*)d_in[17]; const float* tb3 = (const float*)d_in[18];
    const float* tw4 = (const float*)d_in[19]; const float* tb4 = (const float*)d_in[20];
    const float* tw5 = (const float*)d_in[21]; const float* tb5 = (const float*)d_in[22];

    char* ws = (char*)d_ws;
    size_t off = 0;
    auto alloc = [&](size_t bytes) -> void* {
        void* p = ws + off;
        off = (off + bytes + 255) & ~(size_t)255;
        return p;
    };

    float*    h1       = (float*)   alloc((size_t)BATCH * 64 * 32 * 32 * 4);   // 13.1 MB
    float*    h2       = (float*)   alloc((size_t)BATCH * 128 * 16 * 16 * 4);  // 6.6 MB
    float*    h3       = (float*)   alloc((size_t)BATCH * 256 * 8 * 8 * 4);    // 3.3 MB
    float*    h4       = (float*)   alloc((size_t)BATCH * 512 * 4 * 4 * 4);    // 1.6 MB
    float*    mu       = (float*)   alloc(512 * 4);
    float*    rsig     = (float*)   alloc(512 * 4);
    _Float16* Aenc     = (_Float16*)alloc((size_t)64 * 8192 * 2);              // 1 MB
    float*    wepad    = (float*)   alloc((size_t)256 * 8192 * 4);             // 8.4 MB
    float*    e        = (float*)   alloc((size_t)BATCH * 200 * 4);
    float*    vv       = (float*)   alloc((size_t)BATCH * 200 * 4);
    _Float16* Adec     = (_Float16*)alloc((size_t)64 * 40000 * 2);             // 5.1 MB
    float*    part_dec = (float*)   alloc((size_t)10 * 64 * 4608 * 4);         // 11.8 MB
    float*    part_enc = (float*)   alloc((size_t)1 * 64 * 256 * 4);
    float*    dfc      = (float*)   alloc((size_t)BATCH * 4608 * 4);           // 0.9 MB

    // transposed-conv outputs alias finished encoder buffers (sizes verified <=)
    float* t1o = h4;   // 160,000 <= 409,600 floats
    float* t2o = h3;   // 387,200 <= 819,200
    float* t3o = h2;   // 846,400 <= 1,638,400
    float* t4o = h1;   // 1,767,200 <= 3,276,800
    float* y    = (float*)d_out;                 // (50,1,49,49)
    float* Cout = (float*)d_out + 120050;        // (50,200,200)

    const int TB = 256;

    // ---- encoder ----
    int n1 = BATCH * 64 * 32 * 32;
    conv3x3_s2p1_leaky<<<cdiv(n1, TB), TB, 0, stream>>>(x,  cw1, cb1, h1, 1,   64, 64, 64,  32, 32);
    bn_stats<<<64, TB, 0, stream>>>(h1, mu, rsig, 64, 1024);
    bn_apply<<<cdiv(n1, TB), TB, 0, stream>>>(h1, mu, rsig, 64, 1024, n1);

    int n2 = BATCH * 128 * 16 * 16;
    conv3x3_s2p1_leaky<<<cdiv(n2, TB), TB, 0, stream>>>(h1, cw2, cb2, h2, 64,  32, 32, 128, 16, 16);
    bn_stats<<<128, TB, 0, stream>>>(h2, mu, rsig, 128, 256);
    bn_apply<<<cdiv(n2, TB), TB, 0, stream>>>(h2, mu, rsig, 128, 256, n2);

    int n3 = BATCH * 256 * 8 * 8;
    conv3x3_s2p1_leaky<<<cdiv(n3, TB), TB, 0, stream>>>(h2, cw3, cb3, h3, 128, 16, 16, 256, 8, 8);
    bn_stats<<<256, TB, 0, stream>>>(h3, mu, rsig, 256, 64);
    bn_apply<<<cdiv(n3, TB), TB, 0, stream>>>(h3, mu, rsig, 256, 64, n3);

    int n4 = BATCH * 512 * 4 * 4;
    conv3x3_s2p1_leaky<<<cdiv(n4, TB), TB, 0, stream>>>(h3, cw4, cb4, h4, 256, 8, 8, 512, 4, 4);
    bn_stats<<<512, TB, 0, stream>>>(h4, mu, rsig, 512, 16);
    bn_apply<<<cdiv(n4, TB), TB, 0, stream>>>(h4, mu, rsig, 512, 16, n4);

    // ---- encoder FC (WMMA): e = relu(h4 @ we.T + be), M=50->64, N=200->256, K=8192 ----
    pack_rows_f16<<<cdiv(64 * 8192, TB), TB, 0, stream>>>(h4, Aenc, BATCH, 8192);
    pad_rows_f32<<<cdiv(256 * 8192, TB), TB, 0, stream>>>(we, wepad, 200, 256, 8192);
    gemm_nt_wmma<<<dim3(4, 1), TB, 0, stream>>>(Aenc, wepad, part_enc, 8192, 256, 256);
    fc_finalize<<<cdiv(BATCH * 200, TB), TB, 0, stream>>>(part_enc, be, e, 200, 256, 1, 1);

    // ---- real iDFT + circulant C ----
    ifft2_real<<<cdiv(BATCH * 200, TB), TB, 0, stream>>>(e, vv);
    build_C_Adec<<<cdiv(64 * 40000, TB), TB, 0, stream>>>(vv, Adec, Cout);

    // ---- decoder FC (WMMA, dominant): d = C @ wd.T + bd, M=50->64, N=4608, K=40000 ----
    gemm_nt_wmma<<<dim3(72, 10), TB, 0, stream>>>(Adec, wd, part_dec, 40000, 4608, 125);
    fc_finalize<<<cdiv(BATCH * 4608, TB), TB, 0, stream>>>(part_dec, bd, dfc, 4608, 4608, 10, 0);

    // ---- transposed-conv decoder ----
    convt3x3<<<cdiv(BATCH * 128 * 5 * 5, TB), TB, 0, stream>>>(dfc, tw1, tb1, t1o, 512, 3, 3, 128, 5, 5, 2, 1);
    convt3x3<<<cdiv(BATCH * 64 * 11 * 11, TB), TB, 0, stream>>>(t1o, tw2, tb2, t2o, 128, 5, 5, 64, 11, 11, 2, 0);
    convt3x3<<<cdiv(BATCH * 32 * 23 * 23, TB), TB, 0, stream>>>(t2o, tw3, tb3, t3o, 64, 11, 11, 32, 23, 23, 2, 0);
    convt3x3<<<cdiv(BATCH * 16 * 47 * 47, TB), TB, 0, stream>>>(t3o, tw4, tb4, t4o, 32, 23, 23, 16, 47, 47, 2, 0);
    convt3x3<<<cdiv(BATCH * 1 * 49 * 49, TB), TB, 0, stream>>>(t4o, tw5, tb5, y,   16, 47, 47, 1, 49, 49, 1, 0);
}
